// GLIMPSE_12936441496213
// MI455X (gfx1250) — matrix operand order repaired and checked by reading, not compiled, not run on hardware
//
#include <hip/hip_runtime.h>
#include <hip/hip_bf16.h>

// ---------------------------------------------------------------------------
// GLIMPSE kernel for MI455X (gfx1250, wave32, WMMA)
//
// mean_t(relu(h)@W^T + b) == (mean_t relu(h))@W^T + b, and glimpse windows
// are linear gathers -> both post-LSTM reductions are accumulated on the fly
// inside the recurrence; the 256x256 linear is applied once to time-summed
// vectors. No [B,T,H] intermediate ever hits memory.
//
// LSTM recurrence: 8 persistent WGs per LSTM, W_hh slice stationary in VGPRs
// as bf16 WMMA B-fragments, h double-buffered in global (L2), one atomic
// inter-WG barrier per timestep. Activations use CDNA5 V_TANH_F32; h staging
// uses GLOBAL_LOAD_ASYNC_TO_LDS_B128 (ASYNCcnt-tracked DMA into LDS).
// ---------------------------------------------------------------------------

#define LATENT 256
#define SELEN  64
#define MSIZE  8
#define BS     32
#define TT     2048
#define NWG    8      // workgroups per LSTM (h-slice of 32 columns each)
#define HSTR   264    // padded LDS row stride for h (ushort), conflict-free
#define GSTR   132    // padded LDS row stride for gates (float)

#define LDS_SPACE __attribute__((address_space(3)))
#define GLB_SPACE __attribute__((address_space(1)))

typedef __attribute__((ext_vector_type(16))) __bf16 v16bf;
typedef __attribute__((ext_vector_type(8)))  float  v8f;
typedef int v4i_async __attribute__((vector_size(16)));   // matches builtin param

union Frag16 {
  v16bf v;
  unsigned short s[16];
  uint4 q[2];
};

__device__ __forceinline__ unsigned short f2bf(float f) {
  unsigned u = __float_as_uint(f);
  u += 0x7FFFu + ((u >> 16) & 1u);     // round-to-nearest-even
  return (unsigned short)(u >> 16);
}

// CDNA5 hardware tanh (TRANS32 unit -> co-executes with WMMA)
__device__ __forceinline__ float fast_tanh(float x) {
#if __has_builtin(__builtin_amdgcn_tanhf)
  return __builtin_amdgcn_tanhf(x);
#elif __has_builtin(__builtin_amdgcn_tanh_f32)
  return __builtin_amdgcn_tanh_f32(x);
#else
  float r;
  asm("v_tanh_f32 %0, %1" : "=v"(r) : "v"(x));
  return r;
#endif
}

__device__ __forceinline__ float fast_sigmoid(float x) {
  return __builtin_fmaf(fast_tanh(0.5f * x), 0.5f, 0.5f);
}

// workspace layout (bytes):
//   [0,     65536) : h double buffers, bf16-in-ushort [lstm][phase][32][256]
//   [65536, 65792) : inter-WG barrier counters (1 per LSTM, 64B apart)
//   [66048, ...  ) : S sums f32 [lstm][kind(0=full,1=glimpse)][32][256]
#define WS_HB  0
#define WS_BAR 65536
#define WS_S   66048

__global__ void glimpse_init(unsigned* __restrict__ ws_u32) {
  const int n = WS_S / 4;   // zero h buffers + barrier counters
  for (int i = blockIdx.x * blockDim.x + threadIdx.x; i < n;
       i += gridDim.x * blockDim.x)
    ws_u32[i] = 0u;
}

__global__ __launch_bounds__(256) void glimpse_lstm(
    const float* __restrict__ touch_d, const float* __restrict__ gaze_d,
    const float* __restrict__ l,
    const float* __restrict__ tW_ih, const float* __restrict__ tW_hh,
    const float* __restrict__ tb_ih, const float* __restrict__ tb_hh,
    const float* __restrict__ gW_ih, const float* __restrict__ gW_hh,
    const float* __restrict__ gb_ih, const float* __restrict__ gb_hh,
    unsigned char* __restrict__ ws)
{
  const int wg   = blockIdx.x;   // 0..7: owns h columns [32*wg, 32*wg+32)
  const int lstm = blockIdx.y;   // 0=touch, 1=gaze
  const int tid  = threadIdx.x;
  const int wv   = tid >> 5;     // wave id 0..7 == N-tile of our gate slice
  const int ln   = tid & 31;

  const float *x, *W_ih, *W_hh, *b_ih, *b_hh; int IN;
  if (lstm == 0) { x = touch_d; W_ih = tW_ih; W_hh = tW_hh; b_ih = tb_ih; b_hh = tb_hh; IN = 3; }
  else           { x = gaze_d;  W_ih = gW_ih; W_hh = gW_hh; b_ih = gb_ih; b_hh = gb_hh; IN = 4; }

  unsigned short* hb  = (unsigned short*)(ws + WS_HB) + (size_t)lstm * 2 * BS * LATENT;
  unsigned*       bar = (unsigned*)(ws + WS_BAR) + lstm * 16;
  float*          Sb  = (float*)(ws + WS_S) + (size_t)lstm * 2 * BS * LATENT;

  __shared__ __align__(16) unsigned short h_lds[BS * HSTR];   // 16.5 KB
  __shared__ __align__(16) float gate_lds[BS * GSTR];         // 16.5 KB
  __shared__ float x_lds[BS * 4];

  // --- stationary per-lane gate column: n_local in [0,128) over our slice,
  //     layout [i(32) | f(32) | g(32) | o(32)] ---
  const int n_local = wv * 16 + (ln & 15);
  const int gate = n_local >> 5;
  const int jcol = n_local & 31;
  const int wrow = gate * LATENT + wg * 32 + jcol;  // row in [4H, H] weights

  // W_hh B-fragments (32x16 bf16), resident in VGPRs for the whole T loop.
  // B layout: lane L holds N = L&15; K half = (L>>4)*16, 16 contiguous K.
  Frag16 breg[8];
  {
    const float* wr = W_hh + (size_t)wrow * LATENT;
    #pragma unroll
    for (int kc = 0; kc < 8; ++kc) {
      const int ks = kc * 32 + ((ln >> 4) << 4);
      #pragma unroll
      for (int e = 0; e < 16; ++e) breg[kc].s[e] = f2bf(wr[ks + e]);
    }
  }
  float wih[4];
  #pragma unroll
  for (int d = 0; d < 4; ++d) wih[d] = (d < IN) ? W_ih[wrow * IN + d] : 0.0f;
  const float bias_n = b_ih[wrow] + b_hh[wrow];

  // --- per-thread cell state: thread owns (cb, 4 h-columns) for all T ---
  const int cb = tid >> 3;              // batch row 0..31
  const int jb = (tid & 7) * 4;         // local h column base
  float c[4]  = {0,0,0,0};
  float s1[4] = {0,0,0,0};              // sum_t relu(h)
  float s2[4] = {0,0,0,0};              // sum over glimpse windows of relu(h)
  int st[MSIZE];
  #pragma unroll
  for (int m = 0; m < MSIZE; ++m) st[m] = (int)(l[cb * MSIZE + m] * (float)TT);

  unsigned target = NWG;

  for (int t = 0; t < TT; ++t) {
    const int pr = t & 1;
    const unsigned short* hg_r = hb + pr * BS * LATENT;
    unsigned short*       hg_w = hb + (pr ^ 1) * BS * LATENT;

    // ---- stage full h (16KB bf16) into LDS ----
    {
      const int row = tid >> 3, seg = tid & 7;
      const unsigned short* src = hg_r + row * LATENT + seg * 32;
      unsigned short*       dst = h_lds + row * HSTR + seg * 32;
#if __has_builtin(__builtin_amdgcn_global_load_async_to_lds_b128)
      // direct global->LDS DMA path, tracked by ASYNCcnt; no VGPR staging
      #pragma unroll
      for (int q = 0; q < 4; ++q)
        __builtin_amdgcn_global_load_async_to_lds_b128(
            (GLB_SPACE v4i_async*)(src + q * 8),
            (LDS_SPACE v4i_async*)(dst + q * 8), 0, 0);
#else
      const uint4* s4 = (const uint4*)src;
      uint4*       d4 = (uint4*)dst;
      d4[0] = s4[0]; d4[1] = s4[1]; d4[2] = s4[2]; d4[3] = s4[3];
#endif
    }
    // stage x_t; prefetch next step's x row
    if (tid < BS * 4) {
      const int b = tid >> 2, d = tid & 3;
      x_lds[tid] = (d < IN) ? x[((size_t)b * TT + t) * IN + d] : 0.0f;
      if (d == 0 && t + 1 < TT)
        __builtin_prefetch(&x[((size_t)b * TT + (t + 1)) * IN], 0, 0);
    }
#if __has_builtin(__builtin_amdgcn_global_load_async_to_lds_b128)
#if __has_builtin(__builtin_amdgcn_s_wait_asynccnt)
    __builtin_amdgcn_s_wait_asynccnt(0);
#else
    asm volatile("s_wait_asynccnt 0" ::: "memory");
#endif
#endif
    __syncthreads();

    // gates[:, our 16 cols] = h @ W_hh^T : 16 chained bf16 WMMAs per wave
    v8f acc0 = {}; v8f acc1 = {};
    {
      const int arow  = ln & 15;
      const int khalf = (ln >> 4) * 16;  // A layout: lanes<16 K{0..15}
      #pragma unroll
      for (int kc = 0; kc < 8; ++kc) {
        Frag16 a0, a1;
        const int kb = kc * 32 + khalf;
        a0.q[0] = *(const uint4*)(h_lds + arow        * HSTR + kb);
        a0.q[1] = *(const uint4*)(h_lds + arow        * HSTR + kb + 8);
        a1.q[0] = *(const uint4*)(h_lds + (16 + arow) * HSTR + kb);
        a1.q[1] = *(const uint4*)(h_lds + (16 + arow) * HSTR + kb + 8);
        acc0 = __builtin_amdgcn_wmma_f32_16x16x32_bf16(
                   false, a0.v, false, breg[kc].v, (short)0, acc0, false, false);
        acc1 = __builtin_amdgcn_wmma_f32_16x16x32_bf16(
                   false, a1.v, false, breg[kc].v, (short)0, acc1, false, false);
      }
    }

    // epilogue in C layout (VGPR r, lane: M = r + 8*(ln>=16), N = ln&15):
    // add x_t @ W_ih^T + bias, spill pre-activation gates to LDS
    {
      const int hi = (ln >> 4) * 8;
      #pragma unroll
      for (int r = 0; r < 8; ++r) {
        const int b0 = r + hi, b1 = 16 + r + hi;
        float xp0 = bias_n, xp1 = bias_n;
        #pragma unroll
        for (int d = 0; d < 4; ++d) {
          xp0 += x_lds[b0 * 4 + d] * wih[d];
          xp1 += x_lds[b1 * 4 + d] * wih[d];
        }
        gate_lds[b0 * GSTR + n_local] = acc0[r] + xp0;
        gate_lds[b1 * GSTR + n_local] = acc1[r] + xp1;
      }
    }
    __syncthreads();

    // cell update + fused reductions for owned (cb, jb..jb+3)
    int wcov = 0;
    #pragma unroll
    for (int m = 0; m < MSIZE; ++m)
      wcov += (t >= st[m] && (t - st[m]) < SELEN) ? 1 : 0;
    const float wf = (float)wcov;
    #pragma unroll
    for (int u = 0; u < 4; ++u) {
      const int j = jb + u;
      const float ig = gate_lds[cb * GSTR + j];
      const float fg = gate_lds[cb * GSTR + 32 + j];
      const float gg = gate_lds[cb * GSTR + 64 + j];
      const float og = gate_lds[cb * GSTR + 96 + j];
      const float cc = fast_sigmoid(fg) * c[u] + fast_sigmoid(ig) * fast_tanh(gg);
      c[u] = cc;
      const float h = fast_sigmoid(og) * fast_tanh(cc);
      hg_w[cb * LATENT + wg * 32 + j] = f2bf(h);
      const float r = fmaxf(h, 0.0f);
      s1[u] += r;
      s2[u] += wf * r;
    }
    __syncthreads();

    // inter-WG barrier (this LSTM's 8 WGs); double-buffered h -> 1/step
    __threadfence();
    if (tid == 0) {
      __hip_atomic_fetch_add(bar, 1u, __ATOMIC_RELEASE, __HIP_MEMORY_SCOPE_AGENT);
      while (__hip_atomic_load(bar, __ATOMIC_ACQUIRE, __HIP_MEMORY_SCOPE_AGENT) < target)
        __builtin_amdgcn_s_sleep(1);
    }
    target += NWG;
    __syncthreads();
    __threadfence();
  }

  // write time-summed vectors
  #pragma unroll
  for (int u = 0; u < 4; ++u) {
    Sb[cb * LATENT + wg * 32 + jb + u]               = s1[u];  // kind 0
    Sb[BS * LATENT + cb * LATENT + wg * 32 + jb + u] = s2[u];  // kind 1
  }
}

// Tiny epilogue: out[b, 1024] = concat of 4 affine maps of the mean vectors.
// 16 MFLOP total -> plain VALU is fine.
__global__ __launch_bounds__(256) void glimpse_final(
    const float* __restrict__ t_lin_w, const float* __restrict__ t_lin_b,
    const float* __restrict__ g_lin_w, const float* __restrict__ g_lin_b,
    const unsigned char* __restrict__ ws, float* __restrict__ out)
{
  const float* S = (const float*)(ws + WS_S);
  const int idx = blockIdx.x * blockDim.x + threadIdx.x;
  if (idx >= BS * 4 * LATENT) return;
  const int b   = idx >> 10;
  const int col = idx & 1023;
  const int mat = col >> 8;     // 0 t_mean, 1 t_attn, 2 g_mean, 3 g_attn
  const int n   = col & 255;
  const float* w   = ((mat < 2) ? t_lin_w : g_lin_w) + (size_t)n * LATENT;
  const float bias = (mat < 2) ? t_lin_b[n] : g_lin_b[n];
  const float scale = (mat & 1) ? (1.0f / (MSIZE * SELEN)) : (1.0f / TT);
  const float* srow = S + (((size_t)(mat >> 1) * 2 + (mat & 1)) * BS + b) * LATENT;
  const float4* w4 = (const float4*)w;
  const float4* s4 = (const float4*)srow;
  float acc = 0.0f;
  #pragma unroll 4
  for (int k = 0; k < LATENT / 4; ++k) {
    const float4 a = s4[k], bb = w4[k];
    acc += a.x * bb.x + a.y * bb.y + a.z * bb.z + a.w * bb.w;
  }
  out[idx] = bias + acc * scale;
}

extern "C" void kernel_launch(void* const* d_in, const int* in_sizes, int n_in,
                              void* d_out, int out_size, void* d_ws, size_t ws_size,
                              hipStream_t stream) {
  (void)in_sizes; (void)n_in; (void)out_size; (void)ws_size;
  const float* touch_d = (const float*)d_in[0];
  const float* gaze_d  = (const float*)d_in[1];
  const float* l       = (const float*)d_in[2];
  const float* tW_ih   = (const float*)d_in[3];
  const float* tW_hh   = (const float*)d_in[4];
  const float* tb_ih   = (const float*)d_in[5];
  const float* tb_hh   = (const float*)d_in[6];
  const float* gW_ih   = (const float*)d_in[7];
  const float* gW_hh   = (const float*)d_in[8];
  const float* gb_ih   = (const float*)d_in[9];
  const float* gb_hh   = (const float*)d_in[10];
  const float* t_lin_w = (const float*)d_in[11];
  const float* t_lin_b = (const float*)d_in[12];
  const float* g_lin_w = (const float*)d_in[13];
  const float* g_lin_b = (const float*)d_in[14];

  glimpse_init<<<16, 256, 0, stream>>>((unsigned*)d_ws);
  glimpse_lstm<<<dim3(NWG, 2), 256, 0, stream>>>(
      touch_d, gaze_d, l, tW_ih, tW_hh, tb_ih, tb_hh,
      gW_ih, gW_hh, gb_ih, gb_hh, (unsigned char*)d_ws);
  glimpse_final<<<(BS * 4 * LATENT + 255) / 256, 256, 0, stream>>>(
      t_lin_w, t_lin_b, g_lin_w, g_lin_b,
      (const unsigned char*)d_ws, (float*)d_out);
}
